// E60ResidualNonlinear_45792941310474
// MI455X (gfx1250) — matrix operand order repaired and checked by reading, not compiled, not run on hardware
//
#include <hip/hip_runtime.h>
#include <hip/hip_bf16.h>
#include <math.h>

// ---------------------------------------------------------------------------
// MI455X (gfx1250) implementation of the gated residual recurrent block.
//   Phase 1: u  = silu(x @ W_in^T)          bf16 WMMA (v_wmma_f32_16x16x32_bf16)
//   Phase 2: wx = u @ W_x^T + b  ([T,B,D])  bf16 WMMA
//   Phase 3: scan over T: pre = h@W_h^T+wx; h += alpha*tanh(pre); out = h*silu(h)
//            persistent kernel, 128 blocks; W_h slice (128KB) pulled into LDS
//            via the Tensor Data Mover (tensor_load_to_lds + s_wait_tensorcnt);
//            f32 WMMA (v_wmma_f32_16x16x4_f32) keeps h in full f32 across the
//            1024 dependent steps; grid-wide sense-reversing barrier per step.
//   Phase 4: y = outs @ W_out^T             bf16 WMMA
// ---------------------------------------------------------------------------

typedef __attribute__((ext_vector_type(16))) __bf16 v16bf;
typedef __attribute__((ext_vector_type(8)))  float  v8f;
typedef __attribute__((ext_vector_type(2)))  float  v2f;

// TDM descriptor groups (probe-reported builtin parameter types)
typedef unsigned int v4u __attribute__((ext_vector_type(4)));
typedef int          v8i __attribute__((ext_vector_type(8)));
typedef int          v4i __attribute__((ext_vector_type(4)));

#define D_DIM 2048
#define B_DIM 8
#define T_DIM 1024
#define M_DIM (B_DIM * T_DIM)   // 8192 rows for the parallel GEMMs

// ------------------------------- helpers -----------------------------------

__global__ void cast_f32_to_bf16(const float* __restrict__ src,
                                 __bf16* __restrict__ dst, int n) {
  int i = (blockIdx.x * blockDim.x + threadIdx.x) * 8;
  if (i + 8 <= n) {
    float4 a = *(const float4*)(src + i);
    float4 b = *(const float4*)(src + i + 4);
    __bf16 t[8] = {(__bf16)a.x, (__bf16)a.y, (__bf16)a.z, (__bf16)a.w,
                   (__bf16)b.x, (__bf16)b.y, (__bf16)b.z, (__bf16)b.w};
    *(uint4*)(dst + i) = *(const uint4*)t;
  }
}

// h double-buffer: 2 x [16 x D] f32. Rows 0..7 of buffer0 = h0, everything
// else zero (rows 8..15 are the M-padding of the 16x16 WMMA tile and must
// stay zero/finite). Also resets the grid barrier each launch (graph-safe).
__global__ void init_state(const float* __restrict__ h0,
                           float* __restrict__ hbuf,
                           unsigned* __restrict__ bar) {
  int i = blockIdx.x * blockDim.x + threadIdx.x;
  if (i < 2 * 16 * D_DIM)
    hbuf[i] = (i < B_DIM * D_DIM) ? h0[i] : 0.0f;
  if (i == 0) { bar[0] = 0u; bar[1] = 0u; }
}

// byte offset of a generic pointer within the wave's LDS allocation
__device__ __forceinline__ unsigned lds_offset_of(void* p) {
  return (unsigned)(size_t)(__attribute__((address_space(3))) char*)p;
}

// -------------------- bf16 WMMA fragment loaders ---------------------------
// A (16x32 bf16, row-major, ld=lda):
//   lane m (<16): K = 0..7 and 16..23 ; lane m+16: K = 8..15 and 24..31
__device__ __forceinline__ v16bf load_a_frag(const __bf16* A, int lda,
                                             int row0, int kb, int lane) {
  int m = lane & 15;
  int koff = (lane < 16) ? 0 : 8;
  const __bf16* p = A + (size_t)(row0 + m) * lda + kb + koff;
  union { uint4 q[2]; v16bf v; } U;
  U.q[0] = *(const uint4*)(p);        // 8 bf16: K koff..koff+7
  U.q[1] = *(const uint4*)(p + 16);   // 8 bf16: K koff+16..koff+23
  return U.v;
}
// B (32x16 bf16) with B[k,n] = W[n,k] (so C = A @ W^T):
//   lane n (<16): column n, K = 0..7 and 16..23 ; lane n+16: column n, K = 8..15 and 24..31
//   -> each lane reads two 8-bf16 runs of row n of row-major W.
__device__ __forceinline__ v16bf load_b_frag(const __bf16* W, int ldw,
                                             int col0, int kb, int lane) {
  int n = lane & 15;
  int koff = (lane < 16) ? 0 : 8;
  const __bf16* p = W + (size_t)(col0 + n) * ldw + kb + koff;
  union { uint4 q[2]; v16bf v; } U;
  U.q[0] = *(const uint4*)(p);
  U.q[1] = *(const uint4*)(p + 16);
  return U.v;
}

// ------------------------- parallel GEMM kernel ----------------------------
// C[M,N] = A[M,K](bf16) @ W[N,K]^T(bf16), f32 accum. 256 thr = 8 waves laid
// out 2(M) x 4(N); each wave computes a 32x32 tile (2x2 register blocking:
// 4 fragment loads feed 4 independent wmma chains). Block tile 64x128.
// EPI 0: out = bf16(silu(c))                              (u)
// EPI 1: out = f32(c + bias[col]), row (b,t) -> (t,b)     (wx, [T,B,D])
// EPI 2: out = f32(c),             row (t,b) -> (b,t)     (y,  [B,T,D])
template <int EPI>
__device__ __forceinline__ void store_tile(const v8f& acc, int row0, int col0,
                                           const float* __restrict__ bias,
                                           void* __restrict__ out, int lane) {
  const int n = lane & 15;
  const int col = col0 + n;
  const float bv = (EPI == 1) ? bias[col] : 0.0f;
#pragma unroll
  for (int vr = 0; vr < 8; ++vr) {
    const int m = vr + ((lane >= 16) ? 8 : 0);
    const int r = row0 + m;            // global A-row index
    float c = acc[vr];
    if (EPI == 0) {
      float s = c / (1.0f + expf(-c)); // silu
      ((__bf16*)out)[(size_t)r * D_DIM + col] = (__bf16)s;
    } else if (EPI == 1) {
      int b_ = r / T_DIM, t_ = r % T_DIM;       // r = b*T + t
      ((float*)out)[((size_t)t_ * B_DIM + b_) * D_DIM + col] = c + bv;
    } else {
      int t_ = r / B_DIM, b_ = r % B_DIM;       // r = t*B + b
      ((float*)out)[((size_t)b_ * T_DIM + t_) * D_DIM + col] = c;
    }
  }
}

template <int EPI>
__global__ void gemm_wmma(const __bf16* __restrict__ A,
                          const __bf16* __restrict__ W,
                          const float* __restrict__ bias,
                          void* __restrict__ out) {
  const int lane = threadIdx.x & 31;
  const int wave = threadIdx.x >> 5;
  const int row0 = blockIdx.y * 64 + (wave >> 2) * 32;    // 2 M-waves
  const int col0 = blockIdx.x * 128 + (wave & 3) * 32;    // 4 N-waves

  v8f acc00 = {}, acc01 = {}, acc10 = {}, acc11 = {};
#pragma unroll 2
  for (int kb = 0; kb < D_DIM; kb += 32) {
    __builtin_prefetch(W + (size_t)(col0 + (lane & 15)) * D_DIM + kb + 128, 0, 1);
    v16bf a0 = load_a_frag(A, D_DIM, row0,      kb, lane);
    v16bf a1 = load_a_frag(A, D_DIM, row0 + 16, kb, lane);
    v16bf b0 = load_b_frag(W, D_DIM, col0,      kb, lane);
    v16bf b1 = load_b_frag(W, D_DIM, col0 + 16, kb, lane);
    acc00 = __builtin_amdgcn_wmma_f32_16x16x32_bf16(false, a0, false, b0,
                                                    (short)0, acc00, false, false);
    acc01 = __builtin_amdgcn_wmma_f32_16x16x32_bf16(false, a0, false, b1,
                                                    (short)0, acc01, false, false);
    acc10 = __builtin_amdgcn_wmma_f32_16x16x32_bf16(false, a1, false, b0,
                                                    (short)0, acc10, false, false);
    acc11 = __builtin_amdgcn_wmma_f32_16x16x32_bf16(false, a1, false, b1,
                                                    (short)0, acc11, false, false);
  }
  store_tile<EPI>(acc00, row0,      col0,      bias, out, lane);
  store_tile<EPI>(acc01, row0,      col0 + 16, bias, out, lane);
  store_tile<EPI>(acc10, row0 + 16, col0,      bias, out, lane);
  store_tile<EPI>(acc11, row0 + 16, col0 + 16, bias, out, lane);
}

// --------------------------- recurrence kernel -----------------------------
// 128 persistent blocks; block owns output cols [16*bx, 16*bx+16).
// W_h rows for those columns (16 x 2048 f32 = 128KB, contiguous) cached in
// LDS once via the Tensor Data Mover. 8 waves split K=2048 into 256-chunks;
// f32 WMMA K=4; LDS cross-wave reduce; double-buffered h.
__global__ void recurrence(const float* __restrict__ Wh,
                           const float* __restrict__ wx,   // [T,B,D] f32
                           const float* __restrict__ log_alpha,
                           float* __restrict__ hbuf,       // 2 x [16,D] f32
                           __bf16* __restrict__ outs,      // [T,B,D] bf16
                           unsigned* __restrict__ bar) {
  extern __shared__ float lds[];
  float* whs = lds;                  // 16*D_DIM floats: W_h slice
  float* red = lds + 16 * D_DIM;     // 8*256 floats: per-wave partials

  const int tid  = threadIdx.x;
  const int lane = tid & 31;
  const int wave = tid >> 5;
  const int col0 = blockIdx.x * 16;
  const unsigned nblk = gridDim.x;

  // ---- stage W_h rows [col0, col0+16) into LDS -----------------------------
#if __has_builtin(__builtin_amdgcn_tensor_load_to_lds) && \
    __has_builtin(__builtin_amdgcn_s_wait_tensorcnt)
  if (wave == 0) {
    // Tensor DMA descriptor (CDNA5 ISA):
    // 2D tile: tile_dim0 = 2048 (f32), tile_dim1 = 16 rows, stride0 = 2048.
    const unsigned long long ga =
        (unsigned long long)(const void*)(Wh + (size_t)col0 * D_DIM);
    v4u g0;
    g0.x = 1u;                                    // count=1, user mode
    g0.y = lds_offset_of(whs);                    // lds_addr (bytes)
    g0.z = (unsigned)(ga & 0xFFFFFFFFu);          // global_addr[31:0]
    g0.w = (unsigned)((ga >> 32) & 0x01FFFFFFu)   // global_addr[56:32]
           | (2u << 30);                          // type = 2 ("image")
    v8i g1;
    g1[0] = (int)(2u << 16);                      // data_size = 2 (4 bytes)
    g1[1] = (int)((unsigned)(D_DIM & 0xFFFF) << 16);   // tensor_dim0 lo16
    g1[2] = (int)((unsigned)16 << 16);            // dim0 hi16=0 | tensor_dim1=16
    g1[3] = (int)((unsigned)(D_DIM & 0xFFFF) << 16);   // dim1 hi16=0 | tile_dim0
    g1[4] = 16;                                   // tile_dim1=16 | tile_dim2=0
    g1[5] = D_DIM;                                // tensor_dim0_stride lo32
    g1[6] = 0;                                    // stride0 hi16 | stride1 lo16
    g1[7] = 0;                                    // stride1 hi32
    v4i g2 = {0, 0, 0, 0};
    v4i g3 = {0, 0, 0, 0};
#if __clang_major__ >= 23
    v8i g4 = {0, 0, 0, 0, 0, 0, 0, 0};
    __builtin_amdgcn_tensor_load_to_lds(g0, g1, g2, g3, g4, 0);
#else
    __builtin_amdgcn_tensor_load_to_lds(g0, g1, g2, g3, 0);
#endif
    __builtin_amdgcn_s_wait_tensorcnt(0);
  }
#else
  {
    const float4* src = (const float4*)(Wh + (size_t)col0 * D_DIM);
    float4* dst = (float4*)whs;
    for (int i = tid; i < 16 * D_DIM / 4; i += 256) dst[i] = src[i];
  }
#endif
  __syncthreads();

  const float alpha = expf(log_alpha[0]);
  const int mn   = lane & 15;               // row m of A == col n of B here
  const int koff = (lane < 16) ? 0 : 2;     // f32 16x16x4 layout split
  const int kb0  = wave * 256;

  for (int t = 0; t < T_DIM; ++t) {
    const float* hcur = hbuf + (size_t)(t & 1) * 16 * D_DIM;
    float*       hnxt = hbuf + (size_t)((t + 1) & 1) * 16 * D_DIM;

    v8f acc0 = {}, acc1 = {};
#pragma unroll 8
    for (int kk = 0; kk < 256; kk += 8) {
      const int kb = kb0 + kk + koff;
      v2f a0 = *(const v2f*)(hcur + (size_t)mn * D_DIM + kb);
      v2f b0 = *(const v2f*)(whs + mn * D_DIM + kb);
      acc0 = __builtin_amdgcn_wmma_f32_16x16x4_f32(false, a0, false, b0,
                                                   (short)0, acc0, false, false);
      v2f a1 = *(const v2f*)(hcur + (size_t)mn * D_DIM + kb + 4);
      v2f b1 = *(const v2f*)(whs + mn * D_DIM + kb + 4);
      acc1 = __builtin_amdgcn_wmma_f32_16x16x4_f32(false, a1, false, b1,
                                                   (short)0, acc1, false, false);
    }

    // dump 16x16 partial tile for cross-wave K-reduction
#pragma unroll
    for (int vr = 0; vr < 8; ++vr) {
      const int m = vr + ((lane >= 16) ? 8 : 0);
      red[wave * 256 + m * 16 + mn] = acc0[vr] + acc1[vr];
    }
    __syncthreads();

    if (tid < B_DIM * 16) {           // 128 live elements (8 batch x 16 cols)
      const int m = tid >> 4, n = tid & 15;
      float s = 0.0f;
#pragma unroll
      for (int w = 0; w < 8; ++w) s += red[w * 256 + m * 16 + n];
      const float pre = s + wx[((size_t)t * B_DIM + m) * D_DIM + col0 + n];
      const float hp  = hcur[(size_t)m * D_DIM + col0 + n];
      const float hn  = hp + alpha * tanhf(pre);
      const float sg  = hn / (1.0f + expf(-hn));   // silu(hn)
      hnxt[(size_t)m * D_DIM + col0 + n] = hn;
      outs[((size_t)t * B_DIM + m) * D_DIM + col0 + n] = (__bf16)(hn * sg);
    }
    __syncthreads();
    __threadfence();

    // grid-wide sense-reversing barrier (bar[0]=count, bar[1]=generation)
    if (tid == 0) {
      unsigned gen = __atomic_load_n(&bar[1], __ATOMIC_ACQUIRE);
      if (atomicAdd(&bar[0], 1u) == nblk - 1u) {
        bar[0] = 0u;
        __threadfence();
        __atomic_store_n(&bar[1], gen + 1u, __ATOMIC_RELEASE);
      } else {
        while (__atomic_load_n(&bar[1], __ATOMIC_ACQUIRE) == gen)
          __builtin_amdgcn_s_sleep(1);
      }
    }
    __syncthreads();
  }
}

// ------------------------------- launcher ----------------------------------

extern "C" void kernel_launch(void* const* d_in, const int* in_sizes, int n_in,
                              void* d_out, int out_size, void* d_ws, size_t ws_size,
                              hipStream_t stream) {
  const float* x    = (const float*)d_in[0];   // [B,T,D]
  const float* h0   = (const float*)d_in[1];   // [B,D]
  const float* Win  = (const float*)d_in[2];   // [D,D]
  const float* Wx   = (const float*)d_in[3];   // [D,D]
  const float* Wh   = (const float*)d_in[4];   // [D,D]
  const float* bias = (const float*)d_in[5];   // [D]
  const float* la   = (const float*)d_in[6];   // scalar
  const float* Wout = (const float*)d_in[7];   // [D,D]
  float* y = (float*)d_out;                    // [B,T,D]

  // workspace layout (~152 MB)
  char* ws = (char*)d_ws;
  size_t off = 0;
  auto alloc = [&](size_t bytes) -> void* {
    void* p = ws + off;
    off += (bytes + 255) & ~(size_t)255;
    return p;
  };
  __bf16*   Winb  = (__bf16*)alloc((size_t)D_DIM * D_DIM * 2);
  __bf16*   Wxb   = (__bf16*)alloc((size_t)D_DIM * D_DIM * 2);
  __bf16*   Woutb = (__bf16*)alloc((size_t)D_DIM * D_DIM * 2);
  __bf16*   xb    = (__bf16*)alloc((size_t)M_DIM * D_DIM * 2); // dead after ph1
  __bf16*   outs  = xb;                                        // alias (ph3 out)
  __bf16*   ub    = (__bf16*)alloc((size_t)M_DIM * D_DIM * 2);
  float*    wx    = (float*)alloc((size_t)M_DIM * D_DIM * 4);
  float*    hbuf  = (float*)alloc((size_t)2 * 16 * D_DIM * 4);
  unsigned* bar   = (unsigned*)alloc(256);
  (void)in_sizes; (void)n_in; (void)out_size; (void)ws_size;

  const int nW = D_DIM * D_DIM;       // 4M elems
  const int nX = M_DIM * D_DIM;       // 16M elems
  cast_f32_to_bf16<<<nW / 8 / 256, 256, 0, stream>>>(Win, Winb, nW);
  cast_f32_to_bf16<<<nW / 8 / 256, 256, 0, stream>>>(Wx, Wxb, nW);
  cast_f32_to_bf16<<<nW / 8 / 256, 256, 0, stream>>>(Wout, Woutb, nW);
  cast_f32_to_bf16<<<nX / 8 / 256, 256, 0, stream>>>(x, xb, nX);
  init_state<<<(2 * 16 * D_DIM + 255) / 256, 256, 0, stream>>>(h0, hbuf, bar);

  dim3 gg(D_DIM / 128, M_DIM / 64);   // (16, 128) blocks of 256 threads
  gemm_wmma<0><<<gg, 256, 0, stream>>>(xb, Winb, nullptr, ub);     // u
  gemm_wmma<1><<<gg, 256, 0, stream>>>(ub, Wxb, bias, wx);         // wx [T,B,D]

  const size_t lds_bytes = (size_t)(16 * D_DIM + 8 * 256) * sizeof(float); // 136KB
  recurrence<<<D_DIM / 16, 256, lds_bytes, stream>>>(Wh, wx, la, hbuf, outs, bar);

  gemm_wmma<2><<<gg, 256, 0, stream>>>(outs, Woutb, nullptr, y);   // y [B,T,D]
}
